// MultiHeadAttention_65481071401968
// MI455X (gfx1250) — matrix operand order repaired and checked by reading, not compiled, not run on hardware
//
#include <hip/hip_runtime.h>
#include <hip/hip_bf16.h>

#define B_DIM 4
#define S_DIM 2048
#define D_DIM 1024
#define H_DIM 16
#define HD_DIM 64

typedef __bf16 bf16_t;
typedef bf16_t v16bf __attribute__((ext_vector_type(16)));
typedef float  v8f   __attribute__((ext_vector_type(8)));

__device__ __forceinline__ bf16_t f2bf(float f) {
    __hip_bfloat16 h = __float2bfloat16(f);
    bf16_t r;
    __builtin_memcpy(&r, &h, sizeof(r));
    return r;
}

// Build a v16bf fragment from two 16-byte chunks (ds_load_b128 / global_load_b128)
__device__ __forceinline__ v16bf load_frag(const bf16_t* p0, const bf16_t* p1) {
    uint4 a = *(const uint4*)p0;
    uint4 b = *(const uint4*)p1;
    v16bf v;
    __builtin_memcpy(&v, &a, 16);
    __builtin_memcpy(reinterpret_cast<char*>(&v) + 16, &b, 16);
    return v;
}

// Convert 16 fp32 -> 16 bf16 and store as two b128 chunks into LDS
__device__ __forceinline__ void cvt16_to_lds(const float* __restrict__ src, bf16_t* dst) {
    float fv[16];
#pragma unroll
    for (int i = 0; i < 4; ++i) {
        float4 f = ((const float4*)src)[i];
        fv[4*i+0] = f.x; fv[4*i+1] = f.y; fv[4*i+2] = f.z; fv[4*i+3] = f.w;
    }
    bf16_t t[16];
#pragma unroll
    for (int i = 0; i < 16; ++i) t[i] = f2bf(fv[i]);
    uint4 u0, u1;
    __builtin_memcpy(&u0, t, 16);
    __builtin_memcpy(&u1, t + 8, 16);
    *(uint4*)dst = u0;
    *(uint4*)(dst + 8) = u1;
}

#define APITCH 40   // 32 bf16 used + 8 pad -> 80B rows, 16B aligned, conflict-free
#define BPITCH 40

// ---------------------------------------------------------------------------
// Kernel 1: q/k/v projections.  Block tile 256(M) x 64(N), K staged 32 at a
// time in LDS as bf16.  Each of the 8 waves owns a 32x64 sub-tile: 2 A frags
// reuse 4 B frags -> 8 WMMAs per chunk, all fragments preloaded so the
// compiler can issue ds_load_b128 clauses and WMMAs back-to-back.
//   z=0 -> q  [B,H,S,HD] bf16,  z=1 -> k [B,H,S,HD] bf16,  z=2 -> vT [B,H,HD,S]
// ---------------------------------------------------------------------------
__global__ __launch_bounds__(256) void qkv_gemm_kernel(
    const float* __restrict__ x,  const float* __restrict__ Wq,
    const float* __restrict__ Wk, const float* __restrict__ Wv,
    bf16_t* __restrict__ qOut, bf16_t* __restrict__ kOut, bf16_t* __restrict__ vTOut)
{
    __shared__ bf16_t As[256 * APITCH];
    __shared__ bf16_t Bs[64 * BPITCH];

    const int tid  = threadIdx.x;
    const int lane = tid & 31;
    const int wv   = tid >> 5;       // 8 waves
    const int l16  = lane & 15;
    const bool hiL = lane >= 16;

    const int mBase = blockIdx.x * 256;   // over B*S = 8192
    const int nBase = blockIdx.y * 64;    // over D
    const int wsel  = blockIdx.z;
    const float* W  = (wsel == 0) ? Wq : ((wsel == 1) ? Wk : Wv);

    v8f c[2][4] = {};

    const int arow = tid >> 1, ahalf = tid & 1;          // 256 rows x 2 halves, 2 tasks/thread
    const int brow = (tid & 127) >> 1, bhalf = tid & 1;  // 64 rows x 2 halves

    for (int kc = 0; kc < D_DIM; kc += 32) {
        {
            const float* s0 = x + (size_t)(mBase + arow) * D_DIM + kc + ahalf * 16;
            const float* s1 = x + (size_t)(mBase + arow + 128) * D_DIM + kc + ahalf * 16;
            cvt16_to_lds(s0, As + arow * APITCH + ahalf * 16);
            cvt16_to_lds(s1, As + (arow + 128) * APITCH + ahalf * 16);
            if (kc + 32 < D_DIM) {
                __builtin_prefetch(s0 + 32, 0, 1);
                __builtin_prefetch(s1 + 32, 0, 1);
            }
        }
        if (tid < 128) {
            const float* src = W + (size_t)(nBase + brow) * D_DIM + kc + bhalf * 16;
            cvt16_to_lds(src, Bs + brow * BPITCH + bhalf * 16);
            if (kc + 32 < D_DIM) __builtin_prefetch(src + 32, 0, 1);
        }
        __syncthreads();

        // Preload ALL fragments, then issue 8 WMMAs back-to-back.
        const int mloc = wv * 32;
        const bf16_t* a0p = As + (mloc + l16) * APITCH + (hiL ? 8 : 0);
        const bf16_t* a1p = As + (mloc + 16 + l16) * APITCH + (hiL ? 8 : 0);
        v16bf af0 = load_frag(a0p, a0p + 16);   // interleaved A layout
        v16bf af1 = load_frag(a1p, a1p + 16);
        v16bf bfr[4];
#pragma unroll
        for (int t = 0; t < 4; ++t) {
            const bf16_t* bp = Bs + (t * 16 + l16) * BPITCH + (hiL ? 8 : 0);
            bfr[t] = load_frag(bp, bp + 16);     // interleaved B layout
        }
#pragma unroll
        for (int t = 0; t < 4; ++t)
            c[0][t] = __builtin_amdgcn_wmma_f32_16x16x32_bf16(
                false, af0, false, bfr[t], (short)0, c[0][t], false, false);
#pragma unroll
        for (int t = 0; t < 4; ++t)
            c[1][t] = __builtin_amdgcn_wmma_f32_16x16x32_bf16(
                false, af1, false, bfr[t], (short)0, c[1][t], false, false);
        __syncthreads();
    }

    // C tile: lane = N (=output channel e), vgpr r = M row (+8 for hi lanes)
    const int h = nBase >> 6;
#pragma unroll
    for (int part = 0; part < 2; ++part) {
        const int m0 = mBase + wv * 32 + part * 16 + (hiL ? 8 : 0);
        if (wsel < 2) {
            bf16_t* out = (wsel == 0) ? qOut : kOut;
#pragma unroll
            for (int t = 0; t < 4; ++t) {
                const int hd = (nBase & 63) + t * 16 + l16;
#pragma unroll
                for (int r = 0; r < 8; ++r) {
                    const int m = m0 + r;
                    const int b = m >> 11;        // / S
                    const int s = m & (S_DIM - 1);
                    out[(((size_t)(b * H_DIM + h)) * S_DIM + s) * HD_DIM + hd] =
                        f2bf(c[part][t][r]);
                }
            }
        } else {
            // transposed V: lane has fixed hd, 8 consecutive s -> one b128 store
            const int b  = m0 >> 11;
            const int s0 = m0 & (S_DIM - 1);
#pragma unroll
            for (int t = 0; t < 4; ++t) {
                const int hd = (nBase & 63) + t * 16 + l16;
                bf16_t tb[8];
#pragma unroll
                for (int r = 0; r < 8; ++r) tb[r] = f2bf(c[part][t][r]);
                uint4 u; __builtin_memcpy(&u, tb, 16);
                *(uint4*)(vTOut + (((size_t)(b * H_DIM + h)) * HD_DIM + hd) * S_DIM + s0) = u;
            }
        }
    }
}

// ---------------------------------------------------------------------------
// Kernel 2: flash attention per (b,h).  Computes S^T = K·Q^T so softmax rows
// live per-lane and P^T feeds the V^T·P^T WMMA with register-only shuffles.
// 128 threads = 4 waves, each wave owns 16 queries; 32-key tiles via LDS.
// ---------------------------------------------------------------------------
#define KPITCH 72   // 64 bf16 + 8 pad
#define VPITCH 40   // 32 bf16 + 8 pad

__global__ __launch_bounds__(128) void attention_kernel(
    const bf16_t* __restrict__ qB, const bf16_t* __restrict__ kB,
    const bf16_t* __restrict__ vT, bf16_t* __restrict__ attnOut)
{
    __shared__ bf16_t Ks[32 * KPITCH];
    __shared__ bf16_t Vs[64 * VPITCH];

    const int tid  = threadIdx.x;
    const int lane = tid & 31;
    const int wv   = tid >> 5;
    const int l16  = lane & 15;
    const bool hiL = lane >= 16;

    const int bh = blockIdx.y;
    const int b  = bh >> 4, h = bh & 15;
    const int qBase = blockIdx.x * 64 + wv * 16;

    const bf16_t* qHead = qB + (size_t)bh * S_DIM * HD_DIM;
    const bf16_t* kHead = kB + (size_t)bh * S_DIM * HD_DIM;
    const bf16_t* vHead = vT + (size_t)bh * HD_DIM * S_DIM;

    // Q as B-operand (interleaved layout), both lane halves read the same query row
    v16bf qf[2];
#pragma unroll
    for (int cc = 0; cc < 2; ++cc) {
        const bf16_t* p = qHead + (size_t)(qBase + l16) * HD_DIM + cc * 32 + (hiL ? 8 : 0);
        qf[cc] = load_frag(p, p + 16);
    }

    v8f acc[4] = {};
    float mrun = -1e30f, lrun = 0.0f;
    const float scale = 0.125f;  // 1/sqrt(64)

    for (int jb = 0; jb < S_DIM; jb += 32) {
        {   // stage K tile [32 j][64 d]
            const int row = tid >> 2, q4 = tid & 3;
            const bf16_t* src = kHead + (size_t)(jb + row) * HD_DIM + q4 * 16;
            uint4 d0 = ((const uint4*)src)[0];
            uint4 d1 = ((const uint4*)src)[1];
            *(uint4*)(Ks + row * KPITCH + q4 * 16)     = d0;
            *(uint4*)(Ks + row * KPITCH + q4 * 16 + 8) = d1;
        }
        {   // stage V^T tile [64 d][32 j]
            const int row = tid >> 1, hf = tid & 1;
            const bf16_t* src = vHead + (size_t)row * S_DIM + jb + hf * 16;
            uint4 d0 = ((const uint4*)src)[0];
            uint4 d1 = ((const uint4*)src)[1];
            *(uint4*)(Vs + row * VPITCH + hf * 16)     = d0;
            *(uint4*)(Vs + row * VPITCH + hf * 16 + 8) = d1;
        }
        __syncthreads();

        // S^T tiles: preload 4 K fragments, then 4 WMMAs back-to-back.
        v16bf kf[2][2];
#pragma unroll
        for (int js = 0; js < 2; ++js)
#pragma unroll
            for (int cc = 0; cc < 2; ++cc) {
                const bf16_t* p = Ks + (js * 16 + l16) * KPITCH + cc * 32 + (hiL ? 8 : 0);
                kf[js][cc] = load_frag(p, p + 16);
            }
        v8f st[2] = {};
#pragma unroll
        for (int js = 0; js < 2; ++js)
#pragma unroll
            for (int cc = 0; cc < 2; ++cc)
                st[js] = __builtin_amdgcn_wmma_f32_16x16x32_bf16(
                    false, kf[js][cc], false, qf[cc], (short)0, st[js], false, false);

        // online softmax: per-lane 16 values + one xor-16 combine per row
        float vmax = -1e30f;
#pragma unroll
        for (int r = 0; r < 8; ++r) vmax = fmaxf(vmax, fmaxf(st[0][r], st[1][r]));
        vmax *= scale;
        vmax = fmaxf(vmax, __shfl_xor(vmax, 16, 32));
        const float mnew  = fmaxf(mrun, vmax);
        const float alpha = __expf(mrun - mnew);

        float p0[8], p1[8], psum = 0.0f;
#pragma unroll
        for (int r = 0; r < 8; ++r) { p0[r] = __expf(st[0][r] * scale - mnew); psum += p0[r]; }
#pragma unroll
        for (int r = 0; r < 8; ++r) { p1[r] = __expf(st[1][r] * scale - mnew); psum += p1[r]; }
        psum += __shfl_xor(psum, 16, 32);
        lrun = lrun * alpha + psum;
        mrun = mnew;
#pragma unroll
        for (int t = 0; t < 4; ++t)
#pragma unroll
            for (int r = 0; r < 8; ++r) acc[t][r] *= alpha;

        // build P^T B-operand fragment (register-only lane-half swap)
        bf16_t pk[16];
#pragma unroll
        for (int r = 0; r < 8; ++r) {
            const float s0 = __shfl_xor(p0[r], 16, 32);
            const float s1 = __shfl_xor(p1[r], 16, 32);
            pk[r]     = f2bf(hiL ? s1 : p0[r]);   // K halves 0..7
            pk[8 + r] = f2bf(hiL ? p1[r] : s0);   // K halves 8..15
        }
        v16bf pf;
        __builtin_memcpy(&pf, pk, 32);

        // out^T += V^T(16d x 32j) x P^T(32j x 16q): preload 4 V frags, 4 WMMAs
        v16bf vf[4];
#pragma unroll
        for (int t = 0; t < 4; ++t) {
            const bf16_t* p = Vs + (t * 16 + l16) * VPITCH + (hiL ? 16 : 0);
            vf[t] = load_frag(p, p + 8);
        }
#pragma unroll
        for (int t = 0; t < 4; ++t)
            acc[t] = __builtin_amdgcn_wmma_f32_16x16x32_bf16(
                false, vf[t], false, pf, (short)0, acc[t], false, false);
        __syncthreads();
    }

    // out^T C tile: lane = query, vgpr = d -> per-lane contiguous d, b128 stores
    const float inv = 1.0f / lrun;
    const int s = qBase + l16;
#pragma unroll
    for (int t = 0; t < 4; ++t) {
        bf16_t tb[8];
#pragma unroll
        for (int r = 0; r < 8; ++r) tb[r] = f2bf(acc[t][r] * inv);
        uint4 u; __builtin_memcpy(&u, tb, 16);
        const size_t off = ((size_t)(b * S_DIM + s)) * D_DIM + h * 64 + t * 16 + (hiL ? 8 : 0);
        *(uint4*)(attnOut + off) = u;
    }
}

// ---------------------------------------------------------------------------
// Kernel 3: final projection  out = attn @ Wo^T  (bf16 A, cvt-staged Wo, f32 out)
// Same 256x64 tiling as kernel 1.
// ---------------------------------------------------------------------------
__global__ __launch_bounds__(256) void out_gemm_kernel(
    const bf16_t* __restrict__ attn, const float* __restrict__ Wo,
    float* __restrict__ out)
{
    __shared__ bf16_t As[256 * APITCH];
    __shared__ bf16_t Bs[64 * BPITCH];

    const int tid  = threadIdx.x;
    const int lane = tid & 31;
    const int wv   = tid >> 5;
    const int l16  = lane & 15;
    const bool hiL = lane >= 16;

    const int mBase = blockIdx.x * 256;
    const int nBase = blockIdx.y * 64;

    v8f c[2][4] = {};

    const int arow = tid >> 1, ahalf = tid & 1;
    const int brow = (tid & 127) >> 1, bhalf = tid & 1;

    for (int kc = 0; kc < D_DIM; kc += 32) {
        {
            const bf16_t* s0 = attn + (size_t)(mBase + arow) * D_DIM + kc + ahalf * 16;
            const bf16_t* s1 = attn + (size_t)(mBase + arow + 128) * D_DIM + kc + ahalf * 16;
            uint4 u0 = ((const uint4*)s0)[0];
            uint4 u1 = ((const uint4*)s0)[1];
            uint4 u2 = ((const uint4*)s1)[0];
            uint4 u3 = ((const uint4*)s1)[1];
            bf16_t* d0 = As + arow * APITCH + ahalf * 16;
            bf16_t* d1 = As + (arow + 128) * APITCH + ahalf * 16;
            *(uint4*)d0 = u0; *(uint4*)(d0 + 8) = u1;
            *(uint4*)d1 = u2; *(uint4*)(d1 + 8) = u3;
            if (kc + 32 < D_DIM) {
                __builtin_prefetch(s0 + 32, 0, 1);
                __builtin_prefetch(s1 + 32, 0, 1);
            }
        }
        if (tid < 128) {
            const float* src = Wo + (size_t)(nBase + brow) * D_DIM + kc + bhalf * 16;
            cvt16_to_lds(src, Bs + brow * BPITCH + bhalf * 16);
        }
        __syncthreads();

        const int mloc = wv * 32;
        const bf16_t* a0p = As + (mloc + l16) * APITCH + (hiL ? 8 : 0);
        const bf16_t* a1p = As + (mloc + 16 + l16) * APITCH + (hiL ? 8 : 0);
        v16bf af0 = load_frag(a0p, a0p + 16);
        v16bf af1 = load_frag(a1p, a1p + 16);
        v16bf bfr[4];
#pragma unroll
        for (int t = 0; t < 4; ++t) {
            const bf16_t* bp = Bs + (t * 16 + l16) * BPITCH + (hiL ? 8 : 0);
            bfr[t] = load_frag(bp, bp + 16);
        }
#pragma unroll
        for (int t = 0; t < 4; ++t)
            c[0][t] = __builtin_amdgcn_wmma_f32_16x16x32_bf16(
                false, af0, false, bfr[t], (short)0, c[0][t], false, false);
#pragma unroll
        for (int t = 0; t < 4; ++t)
            c[1][t] = __builtin_amdgcn_wmma_f32_16x16x32_bf16(
                false, af1, false, bfr[t], (short)0, c[1][t], false, false);
        __syncthreads();
    }

#pragma unroll
    for (int part = 0; part < 2; ++part) {
        const int m0 = mBase + wv * 32 + part * 16 + (hiL ? 8 : 0);
#pragma unroll
        for (int t = 0; t < 4; ++t) {
            const int e = nBase + t * 16 + l16;
#pragma unroll
            for (int r = 0; r < 8; ++r)
                out[(size_t)(m0 + r) * D_DIM + e] = c[part][t][r];
        }
    }
}

extern "C" void kernel_launch(void* const* d_in, const int* in_sizes, int n_in,
                              void* d_out, int out_size, void* d_ws, size_t ws_size,
                              hipStream_t stream) {
    (void)in_sizes; (void)n_in; (void)out_size; (void)ws_size;
    const float* x  = (const float*)d_in[0];
    const float* Wq = (const float*)d_in[1];
    const float* Wk = (const float*)d_in[2];
    const float* Wv = (const float*)d_in[3];
    const float* Wo = (const float*)d_in[4];
    float* out = (float*)d_out;

    const size_t n = (size_t)B_DIM * H_DIM * S_DIM * HD_DIM;  // 8,388,608 elems
    bf16_t* qb   = (bf16_t*)d_ws;
    bf16_t* kb   = qb + n;
    bf16_t* vTb  = kb + n;
    bf16_t* attn = vTb + n;

    dim3 g2((B_DIM * S_DIM) / 256, D_DIM / 64, 3);
    qkv_gemm_kernel<<<g2, dim3(256), 0, stream>>>(x, Wq, Wk, Wv, qb, kb, vTb);

    dim3 g3(S_DIM / 64, B_DIM * H_DIM);
    attention_kernel<<<g3, dim3(128), 0, stream>>>(qb, kb, vTb, attn);

    dim3 g4((B_DIM * S_DIM) / 256, D_DIM / 64);
    out_gemm_kernel<<<g4, dim3(256), 0, stream>>>(attn, Wo, out);
}
